// Attention_919123001805
// MI455X (gfx1250) — compile-verified, matrix operand order checked by reading, not run on hardware
//
#include <hip/hip_runtime.h>

// ---------------------------------------------------------------------------
// MI455X (gfx1250) attention kernel, bf16 WMMA everywhere, fp32 reductions.
// B=8, N=1024, DIM=768, H=12, D=64. ~55 GFLOP vs ~150MB HBM traffic
// (bias read twice to avoid materializing the 403MB score tensor).
// ---------------------------------------------------------------------------

typedef __attribute__((ext_vector_type(16))) __bf16 v16bf;
typedef __attribute__((ext_vector_type(8)))  __bf16 v8bf;
typedef __attribute__((ext_vector_type(8)))  float  v8f;
typedef __attribute__((ext_vector_type(4)))  float  v4f;

#define B_     8
#define N_     1024
#define DIM_   768
#define H_     12
#define D_     64
#define SCALE_ 0.125f     // D^-0.5
#define EPS_   1e-5f
#define LDSS   1032       // 1024 + 8 pad (floats) per score row

__device__ __forceinline__ v8f zero8() {
  v8f z;
#pragma unroll
  for (int i = 0; i < 8; ++i) z[i] = 0.0f;
  return z;
}

__device__ __forceinline__ v8f wmma_bf16(v16bf a, v16bf b, v8f c) {
  // D = A(16x32 bf16) * B(32x16 bf16) + C(16x16 f32)
  return __builtin_amdgcn_wmma_f32_16x16x32_bf16(
      /*neg_a=*/false, a, /*neg_b=*/false, b,
      /*c_mod=*/(short)0, c, /*reuse_a=*/false, /*reuse_b=*/false);
}

// A/B fragment loader, 16-bit 16x32 layout (ISA 7.12.2):
// lane row/col = lane&15, khalf = lane>>4; elems 0..7 -> k = khalf*8 + 0..7,
// elems 8..15 -> k = 16 + khalf*8 + 0..7. Two contiguous 16B (b128) loads.
__device__ __forceinline__ v16bf load_frag(const __bf16* __restrict__ rowptr,
                                           int kbase, int khalf) {
  v8bf lo = *(const v8bf*)(rowptr + kbase + khalf * 8);
  v8bf hi = *(const v8bf*)(rowptr + kbase + 16 + khalf * 8);
  v16bf f;
#pragma unroll
  for (int i = 0; i < 8; ++i) { f[i] = lo[i]; f[i + 8] = hi[i]; }
  return f;
}

// ---------------------------------------------------------------------------
// fp32 -> bf16 conversion (grid-stride)
// ---------------------------------------------------------------------------
__global__ void cvt_bf16_kernel(const float* __restrict__ src,
                                __bf16* __restrict__ dst, int n) {
  int i = blockIdx.x * blockDim.x + threadIdx.x;
  int stride = gridDim.x * blockDim.x;
  for (; i < n; i += stride) dst[i] = (__bf16)src[i];
}

__global__ void zero_f32_kernel(float* __restrict__ p, int n) {
  int i = blockIdx.x * blockDim.x + threadIdx.x;
  if (i < n) p[i] = 0.0f;
}

// ---------------------------------------------------------------------------
// QV projection: C[8192,1536] = X[8192,768] @ Wqv[1536,768]^T
// wave = one 16x64 output tile (4 C frags, A reused across 4 B frags).
// Epilogue scatters: e<768 -> Q[b,h,n,d] ; e>=768 -> Vt[b,h,d,n] (transposed
// so PV-stage B fragments are contiguous b128 loads).
// ---------------------------------------------------------------------------
__global__ void __launch_bounds__(128) gemm_qv_kernel(
    const __bf16* __restrict__ X, const __bf16* __restrict__ W,
    __bf16* __restrict__ Q, __bf16* __restrict__ Vt) {
  const int lane = threadIdx.x & 31;
  const int wid  = threadIdx.x >> 5;
  const int t    = blockIdx.x * 4 + wid;        // 12288 waves
  const int tn   = t % 24, tm = t / 24;         // 24 n-tiles, 512 m-tiles
  const int m0   = tm * 16, n0 = tn * 64;
  const int l15  = lane & 15, kh = lane >> 4;

  const __bf16* arow = X + (size_t)(m0 + l15) * DIM_;
  v8f acc[4] = {zero8(), zero8(), zero8(), zero8()};

  for (int ks = 0; ks < DIM_; ks += 32) {
    v16bf a = load_frag(arow, ks, kh);
#pragma unroll
    for (int nt = 0; nt < 4; ++nt) {
      v16bf bf = load_frag(W + (size_t)(n0 + nt * 16 + l15) * DIM_, ks, kh);
      acc[nt] = wmma_bf16(a, bf, acc[nt]);
    }
  }

#pragma unroll
  for (int nt = 0; nt < 4; ++nt) {
    const int e = n0 + nt * 16 + l15;
#pragma unroll
    for (int r = 0; r < 8; ++r) {
      const int m  = m0 + kh * 8 + r;           // C layout: m = vgpr + 8*khalf
      const int b  = m >> 10, nq = m & 1023;
      const float val = acc[nt][r];
      if (e < DIM_) {
        const int h = e >> 6, d = e & 63;
        Q[((size_t)(b * H_ + h) * N_ + nq) * D_ + d] = (__bf16)val;
      } else {
        const int e2 = e - DIM_;
        const int h = e2 >> 6, d = e2 & 63;
        Vt[((size_t)(b * H_ + h) * D_ + d) * N_ + nq] = (__bf16)val;
      }
    }
  }
}

// ---------------------------------------------------------------------------
// Pass 1 over scores: s = (q.k^T + bias)*SCALE ; accumulate sum / sumsq per
// head (BatchNorm training statistics). Wave = 16 query rows x 1024 keys.
// ---------------------------------------------------------------------------
__global__ void __launch_bounds__(128) qk_stats_kernel(
    const __bf16* __restrict__ Q, const __bf16* __restrict__ K,
    const float* __restrict__ bias, float* __restrict__ stats) {
  const int lane = threadIdx.x & 31;
  const int wid  = threadIdx.x >> 5;
  const int id   = blockIdx.x * 4 + wid;        // 6144 waves
  const int qt   = id & 63;
  const int h    = (id >> 6) % H_;
  const int b    = id / (64 * H_);
  const int l15  = lane & 15, kh = lane >> 4;

  const __bf16* qrow = Q + ((size_t)(b * H_ + h) * N_ + qt * 16 + l15) * D_;
  const v16bf a0 = load_frag(qrow, 0, kh);
  const v16bf a1 = load_frag(qrow, 32, kh);
  const float* bb = bias + (size_t)h * N_ * N_;

  float sum = 0.0f, ssq = 0.0f;
  for (int kt = 0; kt < 64; ++kt) {
    const __bf16* krow = K + ((size_t)h * N_ + kt * 16 + l15) * D_;
    v16bf b0 = load_frag(krow, 0, kh);
    v16bf b1 = load_frag(krow, 32, kh);
    v8f c = zero8();
    c = wmma_bf16(a0, b0, c);
    c = wmma_bf16(a1, b1, c);
    const int col = kt * 16 + l15;
#pragma unroll
    for (int r = 0; r < 8; ++r) {
      const int m = qt * 16 + kh * 8 + r;
      const float s = (c[r] + bb[(size_t)m * N_ + col]) * SCALE_;
      sum += s;
      ssq += s * s;
    }
  }
#pragma unroll
  for (int off = 16; off >= 1; off >>= 1) {
    sum += __shfl_xor(sum, off, 32);
    ssq += __shfl_xor(ssq, off, 32);
  }
  if (lane == 0) {
    atomicAdd(&stats[h * 2 + 0], sum);
    atomicAdd(&stats[h * 2 + 1], ssq);
  }
}

// Fold BN stats into per-head affine: final = (qk+bias)*SCALE*a + c
__global__ void bn_affine_kernel(const float* __restrict__ stats,
                                 const float* __restrict__ gamma,
                                 const float* __restrict__ beta,
                                 float* __restrict__ affine) {
  const int h = threadIdx.x;
  if (h < H_) {
    const float cnt  = (float)B_ * (float)N_ * (float)N_;
    const float mean = stats[h * 2 + 0] / cnt;
    const float var  = stats[h * 2 + 1] / cnt - mean * mean;
    const float rs   = rsqrtf(var + EPS_);
    const float a    = gamma[h] * rs;
    affine[h * 2 + 0] = a;
    affine[h * 2 + 1] = beta[h] - mean * a;
  }
}

// ---------------------------------------------------------------------------
// Pass 2: recompute scores -> BN affine -> LDS -> softmax -> P@V -> O bf16
// in [B,N,H*D] layout (input of output projection). Wave = 16 query rows.
// 2 waves/block, 129KB dynamic LDS (2 blocks/WGP within 320KB).
// ---------------------------------------------------------------------------
__global__ void __launch_bounds__(64) attn_softmax_pv_kernel(
    const __bf16* __restrict__ Q, const __bf16* __restrict__ K,
    const __bf16* __restrict__ Vt, const float* __restrict__ bias,
    const float* __restrict__ affine, __bf16* __restrict__ O) {
  extern __shared__ float smem[];
  const int lane = threadIdx.x & 31;
  const int wid  = threadIdx.x >> 5;
  float* slab = smem + (size_t)wid * 16 * LDSS;

  const int id  = blockIdx.x * 2 + wid;         // 6144 waves
  const int qt  = id & 63;
  const int h   = (id >> 6) % H_;
  const int b   = id / (64 * H_);
  const int l15 = lane & 15, kh = lane >> 4;

  const float sa = affine[h * 2 + 0] * SCALE_;
  const float sc = affine[h * 2 + 1];

  const __bf16* qrow = Q + ((size_t)(b * H_ + h) * N_ + qt * 16 + l15) * D_;
  const v16bf a0 = load_frag(qrow, 0, kh);
  const v16bf a1 = load_frag(qrow, 32, kh);
  const float* bb = bias + (size_t)h * N_ * N_;

  // ---- phase 1: normalized scores into LDS -------------------------------
  for (int kt = 0; kt < 64; ++kt) {
    const __bf16* krow = K + ((size_t)h * N_ + kt * 16 + l15) * D_;
    v16bf b0 = load_frag(krow, 0, kh);
    v16bf b1 = load_frag(krow, 32, kh);
    v8f c = zero8();
    c = wmma_bf16(a0, b0, c);
    c = wmma_bf16(a1, b1, c);
    const int col = kt * 16 + l15;
#pragma unroll
    for (int r = 0; r < 8; ++r) {
      const int mrow = kh * 8 + r;
      const float s  = fmaf(c[r] + bb[(size_t)(qt * 16 + mrow) * N_ + col], sa, sc);
      slab[mrow * LDSS + col] = s;
    }
  }
  __syncthreads();

  // ---- phase 2: row softmax (lane pair l / l+16 per row) ------------------
  float* sp = slab + l15 * LDSS + kh * 512;
  float mx = -3.4e38f;
  for (int i = 0; i < 128; ++i) {
    v4f t = ((const v4f*)sp)[i];
    mx = fmaxf(mx, fmaxf(fmaxf(t[0], t[1]), fmaxf(t[2], t[3])));
  }
  mx = fmaxf(mx, __shfl_xor(mx, 16, 32));
  float sum = 0.0f;
  for (int i = 0; i < 128; ++i) {
    v4f t = ((const v4f*)sp)[i];
    t[0] = __expf(t[0] - mx); t[1] = __expf(t[1] - mx);
    t[2] = __expf(t[2] - mx); t[3] = __expf(t[3] - mx);
    sum += (t[0] + t[1]) + (t[2] + t[3]);
    ((v4f*)sp)[i] = t;
  }
  sum += __shfl_xor(sum, 16, 32);
  const float inv = 1.0f / sum;                 // per-row; A frag row == l15
  __syncthreads();

  // ---- phase 3: O(16x64) = P(16x1024) @ V(1024x64) ------------------------
  v8f o[4] = {zero8(), zero8(), zero8(), zero8()};
  const float* arow = slab + l15 * LDSS;
  for (int k2 = 0; k2 < 32; ++k2) {
    const int kb = k2 * 32 + kh * 8;
    v4f f0 = *(const v4f*)(arow + kb);
    v4f f1 = *(const v4f*)(arow + kb + 4);
    v4f f2 = *(const v4f*)(arow + kb + 16);
    v4f f3 = *(const v4f*)(arow + kb + 20);
    v16bf a;
#pragma unroll
    for (int i = 0; i < 4; ++i) {
      a[i +  0] = (__bf16)(f0[i] * inv);
      a[i +  4] = (__bf16)(f1[i] * inv);
      a[i +  8] = (__bf16)(f2[i] * inv);
      a[i + 12] = (__bf16)(f3[i] * inv);
    }
#pragma unroll
    for (int nt = 0; nt < 4; ++nt) {
      const __bf16* vrow =
          Vt + ((size_t)(b * H_ + h) * D_ + nt * 16 + l15) * N_;
      v16bf bf = load_frag(vrow, k2 * 32, kh);
      o[nt] = wmma_bf16(a, bf, o[nt]);
    }
  }

#pragma unroll
  for (int nt = 0; nt < 4; ++nt) {
#pragma unroll
    for (int r = 0; r < 8; ++r) {
      const int m = qt * 16 + kh * 8 + r;
      const int d = nt * 16 + l15;
      O[((size_t)b * N_ + m) * DIM_ + h * D_ + d] = (__bf16)o[nt][r];
    }
  }
}

// ---------------------------------------------------------------------------
// Output projection: out[8192,768] = O[8192,768] @ Wp[768,768]^T + b_proj
// ---------------------------------------------------------------------------
__global__ void __launch_bounds__(128) gemm_proj_kernel(
    const __bf16* __restrict__ X, const __bf16* __restrict__ W,
    const float* __restrict__ bproj, float* __restrict__ out) {
  const int lane = threadIdx.x & 31;
  const int wid  = threadIdx.x >> 5;
  const int t    = blockIdx.x * 4 + wid;        // 6144 waves
  const int tn   = t % 12, tm = t / 12;
  const int m0   = tm * 16, n0 = tn * 64;
  const int l15  = lane & 15, kh = lane >> 4;

  const __bf16* arow = X + (size_t)(m0 + l15) * DIM_;
  v8f acc[4] = {zero8(), zero8(), zero8(), zero8()};

  for (int ks = 0; ks < DIM_; ks += 32) {
    v16bf a = load_frag(arow, ks, kh);
#pragma unroll
    for (int nt = 0; nt < 4; ++nt) {
      v16bf bf = load_frag(W + (size_t)(n0 + nt * 16 + l15) * DIM_, ks, kh);
      acc[nt] = wmma_bf16(a, bf, acc[nt]);
    }
  }

#pragma unroll
  for (int nt = 0; nt < 4; ++nt) {
    const int n = n0 + nt * 16 + l15;
    const float bv = bproj[n];
#pragma unroll
    for (int r = 0; r < 8; ++r) {
      const int m = m0 + kh * 8 + r;
      out[(size_t)m * DIM_ + n] = acc[nt][r] + bv;
    }
  }
}

// ---------------------------------------------------------------------------
extern "C" void kernel_launch(void* const* d_in, const int* in_sizes, int n_in,
                              void* d_out, int out_size, void* d_ws, size_t ws_size,
                              hipStream_t stream) {
  (void)in_sizes; (void)n_in; (void)out_size; (void)ws_size;
  const float* x        = (const float*)d_in[0];   // [8,1024,768]
  const float* w_qv     = (const float*)d_in[1];   // [1536,768]
  const float* ext_k    = (const float*)d_in[2];   // [1,12,1024,64]
  const float* ext_bias = (const float*)d_in[3];   // [1,12,1024,1024]
  const float* bn_gamma = (const float*)d_in[4];   // [12]
  const float* bn_beta  = (const float*)d_in[5];   // [12]
  const float* w_proj   = (const float*)d_in[6];   // [768,768]
  const float* b_proj   = (const float*)d_in[7];   // [768]
  float* out = (float*)d_out;                      // [8,1024,768] f32

  char* ws = (char*)d_ws;
  size_t off = 0;
  auto carve = [&](size_t bytes) {
    void* p = ws + off;
    off += (bytes + 255) & ~(size_t)255;
    return p;
  };
  const size_t nX  = (size_t)B_ * N_ * DIM_;       // 6291456
  const size_t nWq = (size_t)2 * DIM_ * DIM_;      // 1179648
  const size_t nK  = (size_t)H_ * N_ * D_;         // 786432
  const size_t nWp = (size_t)DIM_ * DIM_;          // 589824
  const size_t nQV = (size_t)B_ * H_ * N_ * D_;    // 6291456

  __bf16* xb   = (__bf16*)carve(nX  * 2);
  __bf16* wqvb = (__bf16*)carve(nWq * 2);
  __bf16* kb   = (__bf16*)carve(nK  * 2);
  __bf16* wpb  = (__bf16*)carve(nWp * 2);
  __bf16* qb   = (__bf16*)carve(nQV * 2);
  __bf16* vtb  = (__bf16*)carve(nQV * 2);
  __bf16* ob   = (__bf16*)carve(nX  * 2);
  float*  stats  = (float*)carve(64 * 4);          // [h*2]: sum, sumsq
  float*  affine = (float*)carve(64 * 4);          // [h*2]: a, c

  // bf16 staging + stat reset
  cvt_bf16_kernel<<<(int)((nX  + 255) / 256), 256, 0, stream>>>(x, xb, (int)nX);
  cvt_bf16_kernel<<<(int)((nWq + 255) / 256), 256, 0, stream>>>(w_qv, wqvb, (int)nWq);
  cvt_bf16_kernel<<<(int)((nK  + 255) / 256), 256, 0, stream>>>(ext_k, kb, (int)nK);
  cvt_bf16_kernel<<<(int)((nWp + 255) / 256), 256, 0, stream>>>(w_proj, wpb, (int)nWp);
  zero_f32_kernel<<<1, 64, 0, stream>>>(stats, 32);

  // 1) qv projection -> Q[b,h,n,d], Vt[b,h,d,n]
  gemm_qv_kernel<<<3072, 128, 0, stream>>>(xb, wqvb, qb, vtb);
  // 2) BN statistics over scores
  qk_stats_kernel<<<1536, 128, 0, stream>>>(qb, kb, ext_bias, stats);
  bn_affine_kernel<<<1, 32, 0, stream>>>(stats, bn_gamma, bn_beta, affine);
  // 3) scores -> BN -> softmax -> P@V  (2 waves/block, 129KB LDS)
  attn_softmax_pv_kernel<<<3072, 64, 2 * 16 * LDSS * sizeof(float), stream>>>(
      qb, kb, vtb, ext_bias, affine, ob);
  // 4) output projection + bias
  gemm_proj_kernel<<<1536, 128, 0, stream>>>(ob, wpb, b_proj, out);
}
